// MoE_67242007986674
// MI455X (gfx1250) — matrix operand order repaired and checked by reading, not compiled, not run on hardware
//
#include <hip/hip_runtime.h>
#include <hip/hip_bf16.h>

// ---------------------------------------------------------------------------
// MoE (top-2 of 8 experts + shared expert), MI455X / gfx1250.
// bf16 WMMA (v_wmma_f32_16x16x32_bf16) for all six GEMMs; router builds
// per-expert gather lists so routed compute is the sparse 26 GF, not the
// dense 103 GF. Compute-bound (~450 FLOP/byte) => matrix-core path.
// ---------------------------------------------------------------------------

typedef __bf16 bf16;
typedef bf16  v16bf __attribute__((ext_vector_type(16)));
typedef float v8f   __attribute__((ext_vector_type(8)));

#define D_DIM   1024
#define H_DIM   512
#define E_NUM   8
#define N_TOK   4096
#define NTILES  (N_TOK / 16)           // 256 sixteen-token tiles
#define LDS_STR (H_DIM + 8)            // 520 bf16 => 1040 B row stride (16B aligned)

union FragU { v16bf v; uint4 q[2]; };

// A fragment, 16x32 bf16, row-major source (same K order as B):
//   lanes 0-15 : M=lane,    K in {k0..k0+15}
//   lanes 16-31: M=lane-16, K in {k0+16..k0+31}
__device__ inline v16bf load_a_frag(const bf16* row, int k0, int lane) {
    const int hi = (lane >> 4) & 1;
    FragU f;
    f.q[0] = *(const uint4*)(row + k0 + hi * 16);
    f.q[1] = *(const uint4*)(row + k0 + hi * 16 + 8);
    return f.v;
}

// B fragment, 32x16 bf16, from TRANSPOSED weights Wt[N][K] (row n contiguous
// over K). Same contiguous K order as A: lanes 0-15 hold K=k0..k0+15 of column N=lane,
// lanes 16-31 hold K=k0+16..k0+31 of column N=lane-16.
__device__ inline v16bf load_b_frag(const bf16* wt_row, int k0, int lane) {
    const int hi = (lane >> 4) & 1;
    FragU f;
    const bf16* p = wt_row + k0 + hi * 16;
    f.q[0] = *(const uint4*)(p);
    f.q[1] = *(const uint4*)(p + 8);
    return f.v;
}

__device__ inline float silu_f(float g) { return g / (1.0f + __expf(-g)); }

// ---------------------------------------------------------------------------
__global__ void moe_init_kernel(int* counts) {
    if (threadIdx.x < E_NUM) counts[threadIdx.x] = 0;
}

__global__ void cast_bf16_kernel(const float* __restrict__ src,
                                 bf16* __restrict__ dst, int n) {
    int i = blockIdx.x * blockDim.x + threadIdx.x;
    if (i < n) dst[i] = (bf16)src[i];
}

// src [batch, K, C] row-major -> dst [batch, C, K] bf16 (transposed per batch)
__global__ void transpose_cast_kernel(const float* __restrict__ src,
                                      bf16* __restrict__ dst,
                                      int K, int C, int total) {
    int i = blockIdx.x * blockDim.x + threadIdx.x;
    if (i >= total) return;
    int kc  = K * C;
    int b   = i / kc;
    int rem = i - b * kc;
    int k   = rem / C;
    int c   = rem - k * C;
    dst[(size_t)b * kc + (size_t)c * K + k] = (bf16)src[i];
}

// ---------------------------------------------------------------------------
// Router: one wave32 per token. logits = x[n] @ W_g, softmax, top-2,
// atomic-append (token, prob) into per-expert gather lists.
__global__ __launch_bounds__(256) void router_kernel(
    const float* __restrict__ x, const float* __restrict__ Wg,
    int* __restrict__ counts, int* __restrict__ tokens, float* __restrict__ wts)
{
    const int lane = threadIdx.x & 31;
    const int wv   = threadIdx.x >> 5;
    const int n    = blockIdx.x * 8 + wv;
    const float* xr = x + (size_t)n * D_DIM;

    float acc[E_NUM];
#pragma unroll
    for (int e = 0; e < E_NUM; ++e) acc[e] = 0.0f;

    for (int d = lane; d < D_DIM; d += 32) {
        float xv = xr[d];
#pragma unroll
        for (int e = 0; e < E_NUM; ++e) acc[e] += xv * Wg[d * E_NUM + e];
    }
#pragma unroll
    for (int e = 0; e < E_NUM; ++e)
        for (int off = 16; off > 0; off >>= 1)
            acc[e] += __shfl_down(acc[e], off, 32);

    if (lane == 0) {
        float mx = acc[0];
#pragma unroll
        for (int e = 1; e < E_NUM; ++e) mx = fmaxf(mx, acc[e]);
        float p[E_NUM], s = 0.0f;
#pragma unroll
        for (int e = 0; e < E_NUM; ++e) { p[e] = __expf(acc[e] - mx); s += p[e]; }
        float inv = 1.0f / s;
        int i1 = 0;
        for (int e = 1; e < E_NUM; ++e) if (p[e] > p[i1]) i1 = e;
        int i2 = (i1 == 0) ? 1 : 0;
        for (int e = 0; e < E_NUM; ++e) if (e != i1 && p[e] > p[i2]) i2 = e;
        int pick[2] = { i1, i2 };
#pragma unroll
        for (int k = 0; k < 2; ++k) {
            int e    = pick[k];
            int slot = atomicAdd(&counts[e], 1);
            tokens[e * N_TOK + slot] = n;
            wts[e * N_TOK + slot]    = p[e] * inv;   // raw softmax prob (no renorm)
        }
    }
}

// ---------------------------------------------------------------------------
// Shared expert: 16-token tile per block, 8 waves.
// Phase 1: gate/up over D=1024 (each wave: 4 H-column tiles x {g,u}).
// Phase 2: down-proj over H=512 (each wave: 8 D-column tiles). Direct store.
__global__ __launch_bounds__(256) void shared_expert_kernel(
    const bf16* __restrict__ xb,   const bf16* __restrict__ Wsg_t,
    const bf16* __restrict__ Wsu_t,const bf16* __restrict__ Wsd_t,
    float* __restrict__ out)
{
    __shared__ __align__(16) bf16 h_lds[16 * LDS_STR];
    const int lane = threadIdx.x & 31;
    const int wv   = threadIdx.x >> 5;
    const int tile = blockIdx.x;
    const int m    = lane & 15;
    const int mrow = (lane < 16) ? 0 : 8;
    const bf16* arow = xb + (size_t)(tile * 16 + m) * D_DIM;

    v8f accg[4], accu[4];
#pragma unroll
    for (int i = 0; i < 4; ++i) { accg[i] = {}; accu[i] = {}; }
    const int nt0 = wv * 4;

    for (int k0 = 0; k0 < D_DIM; k0 += 32) {
        v16bf a = load_a_frag(arow, k0, lane);
#pragma unroll
        for (int i = 0; i < 4; ++i) {
            int n = (nt0 + i) * 16 + m;
            v16bf bg = load_b_frag(Wsg_t + (size_t)n * D_DIM, k0, lane);
            v16bf bu = load_b_frag(Wsu_t + (size_t)n * D_DIM, k0, lane);
            accg[i] = __builtin_amdgcn_wmma_f32_16x16x32_bf16(
                false, a, false, bg, (short)0, accg[i], false, false);
            accu[i] = __builtin_amdgcn_wmma_f32_16x16x32_bf16(
                false, a, false, bu, (short)0, accu[i], false, false);
        }
    }
#pragma unroll
    for (int i = 0; i < 4; ++i) {
        int n = (nt0 + i) * 16 + m;
#pragma unroll
        for (int r = 0; r < 8; ++r) {
            float h = silu_f(accg[i][r]) * accu[i][r];
            h_lds[(mrow + r) * LDS_STR + n] = (bf16)h;
        }
    }
    __syncthreads();

    v8f acco[8];
#pragma unroll
    for (int i = 0; i < 8; ++i) acco[i] = {};
    const bf16* hrow = h_lds + m * LDS_STR;
    const int dt0 = wv * 8;

    for (int k0 = 0; k0 < H_DIM; k0 += 32) {
        v16bf a = load_a_frag(hrow, k0, lane);
#pragma unroll
        for (int i = 0; i < 8; ++i) {
            int dcol = (dt0 + i) * 16 + m;
            v16bf b = load_b_frag(Wsd_t + (size_t)dcol * H_DIM, k0, lane);
            acco[i] = __builtin_amdgcn_wmma_f32_16x16x32_bf16(
                false, a, false, b, (short)0, acco[i], false, false);
        }
    }
#pragma unroll
    for (int i = 0; i < 8; ++i) {
        int dcol = (dt0 + i) * 16 + m;
#pragma unroll
        for (int r = 0; r < 8; ++r)
            out[(size_t)(tile * 16 + mrow + r) * D_DIM + dcol] = acco[i][r];
    }
}

// ---------------------------------------------------------------------------
// Routed experts: grid = E * NTILES. Each block processes one 16-row gathered
// tile of one expert's token list; empty tiles exit immediately. h is scaled
// by the combine weight before the down-proj; result merged with f32 atomics.
__global__ __launch_bounds__(256) void routed_expert_kernel(
    const bf16* __restrict__ xb,  const bf16* __restrict__ Wg_t,
    const bf16* __restrict__ Wu_t,const bf16* __restrict__ Wd_t,
    const int* __restrict__ counts, const int* __restrict__ tokens,
    const float* __restrict__ wts, float* __restrict__ out)
{
    __shared__ __align__(16) bf16 h_lds[16 * LDS_STR];
    __shared__ int   tok_s[16];
    __shared__ float wt_s[16];

    const int e   = blockIdx.x / NTILES;
    const int t   = blockIdx.x % NTILES;
    const int cnt = counts[e];
    if (t * 16 >= cnt) return;                      // uniform: whole block exits

    if (threadIdx.x < 16) {
        int slot = t * 16 + threadIdx.x;
        bool ok  = slot < cnt;
        tok_s[threadIdx.x] = ok ? tokens[e * N_TOK + slot] : 0;
        wt_s[threadIdx.x]  = ok ? wts[e * N_TOK + slot] : 0.0f;
    }
    __syncthreads();

    const int lane = threadIdx.x & 31;
    const int wv   = threadIdx.x >> 5;
    const int m    = lane & 15;
    const int mrow = (lane < 16) ? 0 : 8;
    const bf16* arow = xb + (size_t)tok_s[m] * D_DIM;
    const bf16* Wg_e = Wg_t + (size_t)e * H_DIM * D_DIM;
    const bf16* Wu_e = Wu_t + (size_t)e * H_DIM * D_DIM;
    const bf16* Wd_e = Wd_t + (size_t)e * D_DIM * H_DIM;

    v8f accg[4], accu[4];
#pragma unroll
    for (int i = 0; i < 4; ++i) { accg[i] = {}; accu[i] = {}; }
    const int nt0 = wv * 4;

    for (int k0 = 0; k0 < D_DIM; k0 += 32) {
        v16bf a = load_a_frag(arow, k0, lane);
#pragma unroll
        for (int i = 0; i < 4; ++i) {
            int n = (nt0 + i) * 16 + m;
            v16bf bg = load_b_frag(Wg_e + (size_t)n * D_DIM, k0, lane);
            v16bf bu = load_b_frag(Wu_e + (size_t)n * D_DIM, k0, lane);
            accg[i] = __builtin_amdgcn_wmma_f32_16x16x32_bf16(
                false, a, false, bg, (short)0, accg[i], false, false);
            accu[i] = __builtin_amdgcn_wmma_f32_16x16x32_bf16(
                false, a, false, bu, (short)0, accu[i], false, false);
        }
    }
#pragma unroll
    for (int i = 0; i < 4; ++i) {
        int n = (nt0 + i) * 16 + m;
#pragma unroll
        for (int r = 0; r < 8; ++r) {
            float h = silu_f(accg[i][r]) * accu[i][r] * wt_s[mrow + r];
            h_lds[(mrow + r) * LDS_STR + n] = (bf16)h;
        }
    }
    __syncthreads();

    v8f acco[8];
#pragma unroll
    for (int i = 0; i < 8; ++i) acco[i] = {};
    const bf16* hrow = h_lds + m * LDS_STR;
    const int dt0 = wv * 8;

    for (int k0 = 0; k0 < H_DIM; k0 += 32) {
        v16bf a = load_a_frag(hrow, k0, lane);
#pragma unroll
        for (int i = 0; i < 8; ++i) {
            int dcol = (dt0 + i) * 16 + m;
            v16bf b = load_b_frag(Wd_e + (size_t)dcol * H_DIM, k0, lane);
            acco[i] = __builtin_amdgcn_wmma_f32_16x16x32_bf16(
                false, a, false, b, (short)0, acco[i], false, false);
        }
    }
#pragma unroll
    for (int i = 0; i < 8; ++i) {
        int dcol = (dt0 + i) * 16 + m;
#pragma unroll
        for (int r = 0; r < 8; ++r) {
            int row = mrow + r;
            atomicAdd(&out[(size_t)tok_s[row] * D_DIM + dcol], acco[i][r]);
        }
    }
}

// ---------------------------------------------------------------------------
extern "C" void kernel_launch(void* const* d_in, const int* in_sizes, int n_in,
                              void* d_out, int out_size, void* d_ws, size_t ws_size,
                              hipStream_t stream) {
    (void)in_sizes; (void)n_in; (void)out_size; (void)ws_size;
    const float* x    = (const float*)d_in[0];   // [N, D]
    const float* W_g  = (const float*)d_in[1];   // [D, E]
    const float* W_gt = (const float*)d_in[2];   // [E, D, H]
    const float* W_up = (const float*)d_in[3];   // [E, D, H]
    const float* W_dn = (const float*)d_in[4];   // [E, H, D]
    const float* W_sg = (const float*)d_in[5];   // [D, H]
    const float* W_su = (const float*)d_in[6];   // [D, H]
    const float* W_sd = (const float*)d_in[7];   // [H, D]
    float* out = (float*)d_out;

    char* ws = (char*)d_ws;
    size_t off = 0;
    auto carve = [&](size_t bytes) -> void* {
        void* p = ws + off;
        off += (bytes + 255) & ~(size_t)255;
        return p;
    };
    bf16* xb    = (bf16*)carve((size_t)N_TOK * D_DIM * sizeof(bf16));
    bf16* Wsg_t = (bf16*)carve((size_t)H_DIM * D_DIM * sizeof(bf16));
    bf16* Wsu_t = (bf16*)carve((size_t)H_DIM * D_DIM * sizeof(bf16));
    bf16* Wsd_t = (bf16*)carve((size_t)D_DIM * H_DIM * sizeof(bf16));
    bf16* Wg_t  = (bf16*)carve((size_t)E_NUM * H_DIM * D_DIM * sizeof(bf16));
    bf16* Wu_t  = (bf16*)carve((size_t)E_NUM * H_DIM * D_DIM * sizeof(bf16));
    bf16* Wd_t  = (bf16*)carve((size_t)E_NUM * D_DIM * H_DIM * sizeof(bf16));
    int*   counts = (int*)carve(E_NUM * sizeof(int));
    int*   tokens = (int*)carve((size_t)E_NUM * N_TOK * sizeof(int));
    float* wts    = (float*)carve((size_t)E_NUM * N_TOK * sizeof(float));

    const int thr = 256;
    moe_init_kernel<<<1, 32, 0, stream>>>(counts);

    // fp32 -> bf16 operand prep (one pass, ~70 MB at 23.3 TB/s ~ 6 us)
    cast_bf16_kernel<<<(N_TOK * D_DIM + thr - 1) / thr, thr, 0, stream>>>(
        x, xb, N_TOK * D_DIM);
    transpose_cast_kernel<<<(D_DIM * H_DIM + thr - 1) / thr, thr, 0, stream>>>(
        W_sg, Wsg_t, D_DIM, H_DIM, D_DIM * H_DIM);
    transpose_cast_kernel<<<(D_DIM * H_DIM + thr - 1) / thr, thr, 0, stream>>>(
        W_su, Wsu_t, D_DIM, H_DIM, D_DIM * H_DIM);
    transpose_cast_kernel<<<(H_DIM * D_DIM + thr - 1) / thr, thr, 0, stream>>>(
        W_sd, Wsd_t, H_DIM, D_DIM, H_DIM * D_DIM);
    transpose_cast_kernel<<<(E_NUM * D_DIM * H_DIM + thr - 1) / thr, thr, 0, stream>>>(
        W_gt, Wg_t, D_DIM, H_DIM, E_NUM * D_DIM * H_DIM);
    transpose_cast_kernel<<<(E_NUM * D_DIM * H_DIM + thr - 1) / thr, thr, 0, stream>>>(
        W_up, Wu_t, D_DIM, H_DIM, E_NUM * D_DIM * H_DIM);
    transpose_cast_kernel<<<(E_NUM * H_DIM * D_DIM + thr - 1) / thr, thr, 0, stream>>>(
        W_dn, Wd_t, H_DIM, D_DIM, E_NUM * H_DIM * D_DIM);

    // router -> per-expert gather lists
    router_kernel<<<N_TOK / 8, thr, 0, stream>>>(x, W_g, counts, tokens, wts);

    // shared expert writes out directly (clears poison), then routed experts
    // accumulate on top via f32 atomics (stream order guarantees precedence)
    shared_expert_kernel<<<NTILES, thr, 0, stream>>>(xb, Wsg_t, Wsu_t, Wsd_t, out);
    routed_expert_kernel<<<E_NUM * NTILES, thr, 0, stream>>>(
        xb, Wg_t, Wu_t, Wd_t, counts, tokens, wts, out);
}
